// VisionTransformerEncoder_32856499815172
// MI455X (gfx1250) — compile-verified, matrix operand order checked by reading, not run on hardware
//
#include <hip/hip_runtime.h>
#include <hip/hip_bf16.h>
#include <math.h>

typedef __bf16 bf16_t;
typedef __attribute__((ext_vector_type(16))) __bf16 v16bf;
typedef __attribute__((ext_vector_type(8)))  __bf16 v8bf;
typedef __attribute__((ext_vector_type(8)))  float  v8f;

#define D_MODEL 768
#define NHEAD   12
#define HDIM    64
#define LAYERS  12
#define HID_DIM 3072
#define NPATCH  196
#define BATCH   32
#define ROWS    (BATCH * NPATCH)   /* 6272 = 49*128 */
#define QKV3    (3 * D_MODEL)      /* 2304 */

__device__ __forceinline__ v8f v8f_zero() {
    v8f z;
#pragma unroll
    for (int i = 0; i < 8; ++i) z[i] = 0.0f;
    return z;
}

// 16x32 bf16 WMMA operand fragment from row-major memory (ld in elements).
// ISA 16-bit A layout: lanes 0-15 -> row = lane, K 0-7 / 16-23; lanes 16-31 ->
// same rows, K 8-15 / 24-31. B is staged transposed [n][k] -> same loader.
__device__ __forceinline__ v16bf load_frag(const bf16_t* p, int ld) {
    const int lane = threadIdx.x & 31;
    const bf16_t* q = p + (size_t)(lane & 15) * ld + ((lane >> 4) << 3);
    v8bf lo = *(const v8bf*)(q);
    v8bf hi = *(const v8bf*)(q + 16);
    v16bf f;
#pragma unroll
    for (int e = 0; e < 8; ++e) { f[e] = lo[e]; f[e + 8] = hi[e]; }
    return f;
}

__device__ __forceinline__ v16bf load_frag_rclamp(const bf16_t* p, size_t ld, int rmax) {
    const int lane = threadIdx.x & 31;
    int r = lane & 15;
    if (r > rmax) r = rmax;
    const bf16_t* q = p + (size_t)r * ld + ((lane >> 4) << 3);
    v8bf lo = *(const v8bf*)(q);
    v8bf hi = *(const v8bf*)(q + 16);
    v16bf f;
#pragma unroll
    for (int e = 0; e < 8; ++e) { f[e] = lo[e]; f[e + 8] = hi[e]; }
    return f;
}

__device__ __forceinline__ v8f wmma_bf16(v16bf a, v16bf b, v8f c) {
    return __builtin_amdgcn_wmma_f32_16x16x32_bf16(false, a, false, b, (short)0, c, false, false);
}

// CDNA5 async global->LDS copy (16 bytes per lane), tracked by ASYNCcnt.
__device__ __forceinline__ void async_ld16(unsigned lds_off, const bf16_t* gptr) {
    asm volatile("global_load_async_to_lds_b128 %0, %1, off"
                 :: "v"(lds_off), "v"((unsigned long long)(uintptr_t)gptr)
                 : "memory");
}
__device__ __forceinline__ void wait_async0() {
    asm volatile("s_wait_asynccnt 0x0" ::: "memory");
}

// ---------------------------------------------------------------------------
// Tiled WMMA GEMM: C[M,N] = A[M,K](bf16) @ W(bf16, pre-transposed [N][K]) + bias
// MODE 0: Cf = v   1: Cf += v   2: Cb = bf16(v)   3: Cb = bf16(gelu(v))
// 128x128x32 tile, 8 waves, wave = 32x64 (2x4 accumulators).
// Double-buffered LDS fed by GLOBAL_LOAD_ASYNC_TO_LDS_B128 (ASYNCcnt pipeline):
// issue tile k+1, WMMA tile k, s_wait_asynccnt 0, barrier. One barrier / step.
// M,N,K are multiples of the tile -> no bounds checks.
// ---------------------------------------------------------------------------
template <int MODE>
__global__ __launch_bounds__(256) void gemm_wmma(
    const bf16_t* __restrict__ A, const bf16_t* __restrict__ Bt,
    const float* __restrict__ bias, float* __restrict__ Cf,
    bf16_t* __restrict__ Cb, int M, int N, int K)
{
    __shared__ __align__(16) bf16_t As[2][128][32];
    __shared__ __align__(16) bf16_t Bs[2][128][32];
    const int m0 = blockIdx.y * 128;
    const int n0 = blockIdx.x * 128;
    const int tid = threadIdx.x;
    const int wid = tid >> 5;
    const int lane = tid & 31;
    const int wm = (wid >> 1) * 32;
    const int wn = (wid & 1) * 64;

    // staging geometry: 2 threads per row, 16 elements (32B) each -> 2 async b128
    const int srow = tid >> 1;
    const int sc0 = (tid & 1) * 16;
    const bf16_t* Abase = A + (size_t)(m0 + srow) * K + sc0;
    const bf16_t* Bbase = Bt + (size_t)(n0 + srow) * K + sc0;

    v8f acc[2][4];
#pragma unroll
    for (int i = 0; i < 2; ++i)
#pragma unroll
        for (int j = 0; j < 4; ++j) acc[i][j] = v8f_zero();

    // prologue: fill buffer 0
    async_ld16((unsigned)(uintptr_t)&As[0][srow][sc0], Abase);
    async_ld16((unsigned)(uintptr_t)&As[0][srow][sc0 + 8], Abase + 8);
    async_ld16((unsigned)(uintptr_t)&Bs[0][srow][sc0], Bbase);
    async_ld16((unsigned)(uintptr_t)&Bs[0][srow][sc0 + 8], Bbase + 8);
    wait_async0();
    __syncthreads();

    const int KT = K >> 5;
    for (int kt = 0; kt < KT; ++kt) {
        const int cur = kt & 1;
        if (kt + 1 < KT) {   // async-fill the other buffer
            const int nxt = cur ^ 1;
            const bf16_t* an = Abase + (kt + 1) * 32;
            const bf16_t* bn = Bbase + (kt + 1) * 32;
            async_ld16((unsigned)(uintptr_t)&As[nxt][srow][sc0], an);
            async_ld16((unsigned)(uintptr_t)&As[nxt][srow][sc0 + 8], an + 8);
            async_ld16((unsigned)(uintptr_t)&Bs[nxt][srow][sc0], bn);
            async_ld16((unsigned)(uintptr_t)&Bs[nxt][srow][sc0 + 8], bn + 8);
        }
        if (kt + 3 < KT) {   // pull the weight stream into GL2 early
            __builtin_prefetch((const void*)(Bbase + (kt + 3) * 32), 0, 1);
            __builtin_prefetch((const void*)(Abase + (kt + 3) * 32), 0, 1);
        }
        v16bf af[2], bfr[4];
#pragma unroll
        for (int i = 0; i < 2; ++i) af[i] = load_frag(&As[cur][wm + i * 16][0], 32);
#pragma unroll
        for (int j = 0; j < 4; ++j) bfr[j] = load_frag(&Bs[cur][wn + j * 16][0], 32);
#pragma unroll
        for (int i = 0; i < 2; ++i)
#pragma unroll
            for (int j = 0; j < 4; ++j) acc[i][j] = wmma_bf16(af[i], bfr[j], acc[i][j]);
        if (kt + 1 < KT) wait_async0();
        __syncthreads();
    }

    // Epilogue. C layout: lanes 0-15 -> N=lane, M=r; lanes 16-31 -> N=lane-16, M=r+8.
    const int cn = lane & 15;
    const int half = lane >> 4;
#pragma unroll
    for (int i = 0; i < 2; ++i)
#pragma unroll
        for (int j = 0; j < 4; ++j)
#pragma unroll
            for (int r = 0; r < 8; ++r) {
                const int row = m0 + wm + i * 16 + r + half * 8;
                const int col = n0 + wn + j * 16 + cn;
                const size_t idx = (size_t)row * N + col;
                float v = acc[i][j][r] + bias[col];
                if (MODE == 0) {
                    Cf[idx] = v;
                } else if (MODE == 1) {
                    Cf[idx] += v;
                } else if (MODE == 2) {
                    Cb[idx] = (bf16_t)v;
                } else {
                    const float g = 0.5f * v * (1.0f + erff(v * 0.70710678118654752f));
                    Cb[idx] = (bf16_t)g;
                }
            }
}

// ---------------------------------------------------------------------------
// LayerNorm over D=768, one block per row, 256 threads x 3 elements.
// ---------------------------------------------------------------------------
template <bool BF16OUT>
__global__ __launch_bounds__(256) void layernorm_k(
    const float* __restrict__ x, const float* __restrict__ w, const float* __restrict__ b,
    bf16_t* __restrict__ yb, float* __restrict__ yf)
{
    __shared__ float red[256];
    const int row = blockIdx.x;
    const int tid = threadIdx.x;
    const float* xr = x + (size_t)row * D_MODEL;
    float lx[3];
    float s = 0.0f;
#pragma unroll
    for (int i = 0; i < 3; ++i) { lx[i] = xr[tid + i * 256]; s += lx[i]; }
    red[tid] = s;
    __syncthreads();
    for (int st = 128; st > 0; st >>= 1) {
        if (tid < st) red[tid] += red[tid + st];
        __syncthreads();
    }
    const float mu = red[0] * (1.0f / D_MODEL);
    __syncthreads();
    float v = 0.0f;
#pragma unroll
    for (int i = 0; i < 3; ++i) { const float d = lx[i] - mu; v += d * d; }
    red[tid] = v;
    __syncthreads();
    for (int st = 128; st > 0; st >>= 1) {
        if (tid < st) red[tid] += red[tid + st];
        __syncthreads();
    }
    const float inv = rsqrtf(red[0] * (1.0f / D_MODEL) + 1e-5f);
#pragma unroll
    for (int i = 0; i < 3; ++i) {
        const int c = tid + i * 256;
        const float o = (lx[i] - mu) * inv * w[c] + b[c];
        if (BF16OUT) yb[(size_t)row * D_MODEL + c] = (bf16_t)o;
        else         yf[(size_t)row * D_MODEL + c] = o;
    }
}

// ---------------------------------------------------------------------------
// Attention: one block per (batch, head), 128 threads = 4 waves.
// K [208x64] and V^T [64x224] in LDS (zero padded). Each wave owns a 16-query
// tile: S = Q K^T (2 WMMA x 13 key tiles), in-register half-wave softmax,
// probs -> LDS bf16, O = P V via 7x4 WMMA.
// ---------------------------------------------------------------------------
__global__ __launch_bounds__(128) void attn_k(const bf16_t* __restrict__ qkv,
                                              bf16_t* __restrict__ o)
{
    __shared__ __align__(16) bf16_t Ks[208][64];
    __shared__ __align__(16) bf16_t Vt[64][224];
    __shared__ __align__(16) bf16_t Pw[4][16][224];
    const int bh = blockIdx.x;
    const int b = bh / NHEAD;
    const int h = bh % NHEAD;
    const int tid = threadIdx.x;
    const size_t base = (size_t)b * NPATCH * QKV3 + (size_t)h * HDIM;

    for (int idx = tid; idx < 208 * 64; idx += 128) {
        const int i = idx >> 6, d = idx & 63;
        bf16_t kk = (bf16_t)0.0f, vv = (bf16_t)0.0f;
        if (i < NPATCH) {
            const bf16_t* p = qkv + base + (size_t)i * QKV3 + d;
            kk = p[D_MODEL];
            vv = p[2 * D_MODEL];
        }
        Ks[i][d] = kk;
        Vt[d][i] = vv;
    }
    for (int idx = tid; idx < 64 * 16; idx += 128)
        Vt[idx >> 4][208 + (idx & 15)] = (bf16_t)0.0f;
    for (int idx = tid; idx < 4 * 16 * 224; idx += 128)
        ((bf16_t*)Pw)[idx] = (bf16_t)0.0f;
    __syncthreads();

    const int wid = tid >> 5;
    const int lane = tid & 31;
    const int cn = lane & 15;
    const int half = lane >> 4;
    const bf16_t* qbase = qkv + base;

    for (int qt = wid; qt < 13; qt += 4) {
        const int q0 = qt * 16;
        const int rmax = (NPATCH - 1) - q0;
        v16bf aq0 = load_frag_rclamp(qbase + (size_t)q0 * QKV3, QKV3, rmax);
        v16bf aq1 = load_frag_rclamp(qbase + (size_t)q0 * QKV3 + 32, QKV3, rmax);

        v8f s[13];
#pragma unroll
        for (int kt = 0; kt < 13; ++kt) {
            v8f c = v8f_zero();
            v16bf b0 = load_frag(&Ks[kt * 16][0], 64);
            v16bf b1 = load_frag(&Ks[kt * 16][32], 64);
            c = wmma_bf16(aq0, b0, c);
            c = wmma_bf16(aq1, b1, c);
            s[kt] = c;
        }
#pragma unroll
        for (int kt = 0; kt < 13; ++kt) {
            const int key = kt * 16 + cn;
#pragma unroll
            for (int r = 0; r < 8; ++r)
                s[kt][r] = (key < NPATCH) ? s[kt][r] * 0.125f : -1e30f;
        }
#pragma unroll
        for (int r = 0; r < 8; ++r) {
            float m = -1e30f;
#pragma unroll
            for (int kt = 0; kt < 13; ++kt) m = fmaxf(m, s[kt][r]);
#pragma unroll
            for (int off = 1; off < 16; off <<= 1) m = fmaxf(m, __shfl_xor(m, off, 32));
            float sum = 0.0f;
#pragma unroll
            for (int kt = 0; kt < 13; ++kt) {
                const float e = __expf(s[kt][r] - m);
                s[kt][r] = e;
                sum += e;
            }
#pragma unroll
            for (int off = 1; off < 16; off <<= 1) sum += __shfl_xor(sum, off, 32);
            const float inv = 1.0f / sum;
#pragma unroll
            for (int kt = 0; kt < 13; ++kt) s[kt][r] *= inv;
        }
#pragma unroll
        for (int kt = 0; kt < 13; ++kt) {
            const int key = kt * 16 + cn;
#pragma unroll
            for (int r = 0; r < 8; ++r)
                Pw[wid][r + half * 8][key] = (bf16_t)s[kt][r];
        }
        asm volatile("s_wait_dscnt 0x0" ::: "memory");  // same-wave LDS RAW

        v8f oacc[4];
#pragma unroll
        for (int j = 0; j < 4; ++j) oacc[j] = v8f_zero();
#pragma unroll
        for (int ks = 0; ks < 7; ++ks) {
            v16bf ap = load_frag(&Pw[wid][0][ks * 32], 224);
#pragma unroll
            for (int nt = 0; nt < 4; ++nt) {
                v16bf bv = load_frag(&Vt[nt * 16][ks * 32], 224);
                oacc[nt] = wmma_bf16(ap, bv, oacc[nt]);
            }
        }
#pragma unroll
        for (int nt = 0; nt < 4; ++nt)
#pragma unroll
            for (int r = 0; r < 8; ++r) {
                const int qrow = q0 + r + half * 8;
                if (qrow < NPATCH) {
                    const int dcol = nt * 16 + cn;
                    o[(size_t)(b * NPATCH + qrow) * D_MODEL + h * HDIM + dcol] =
                        (bf16_t)oacc[nt][r];
                }
            }
    }
}

// ---------------------------------------------------------------------------
// Elementwise helpers
// ---------------------------------------------------------------------------
__global__ void patch_gather_k(const float* __restrict__ x, bf16_t* __restrict__ xp) {
    const int idx = blockIdx.x * blockDim.x + threadIdx.x;
    if (idx >= ROWS * D_MODEL) return;
    const int dcol = idx % D_MODEL;
    const int mp = idx / D_MODEL;
    const int p = mp % NPATCH;
    const int b = mp / NPATCH;
    const int c = dcol >> 8;
    const int rem = dcol & 255;
    const int py = rem >> 4, px = rem & 15;
    const int gy = p / 14, gx = p % 14;
    const float v = x[(((size_t)b * 3 + c) * 224 + (gy * 16 + py)) * 224 + gx * 16 + px];
    xp[idx] = (bf16_t)v;
}

// W [K][N] f32 -> Wt [N][K] bf16 (pre-transpose for GEMM B staging)
__global__ void wtrans_cvt_k(const float* __restrict__ w, bf16_t* __restrict__ wt,
                             int K, int N) {
    const int idx = blockIdx.x * blockDim.x + threadIdx.x;
    if (idx >= K * N) return;
    const int n = idx / K, k = idx % K;
    wt[idx] = (bf16_t)w[(size_t)k * N + n];
}

// straight f32 -> bf16 convert (conv_w is already [n][k])
__global__ void cvt_bf16_k(const float* __restrict__ w, bf16_t* __restrict__ o, int n) {
    const int idx = blockIdx.x * blockDim.x + threadIdx.x;
    if (idx < n) o[idx] = (bf16_t)w[idx];
}

__global__ void add_pos_k(float* __restrict__ h, const float* __restrict__ pos) {
    const int idx = blockIdx.x * blockDim.x + threadIdx.x;
    if (idx >= ROWS * D_MODEL) return;
    const int dcol = idx % D_MODEL;
    const int p = (idx / D_MODEL) % NPATCH;
    h[idx] += pos[(size_t)p * D_MODEL + dcol];
}

// ---------------------------------------------------------------------------
extern "C" void kernel_launch(void* const* d_in, const int* in_sizes, int n_in,
                              void* d_out, int out_size, void* d_ws, size_t ws_size,
                              hipStream_t stream) {
    (void)in_sizes; (void)n_in; (void)out_size; (void)ws_size;
    const float* x      = (const float*)d_in[0];
    const float* conv_w = (const float*)d_in[1];
    const float* conv_b = (const float*)d_in[2];
    const float* pos    = (const float*)d_in[3];
    const float* ln1_w  = (const float*)d_in[4];
    const float* ln1_b  = (const float*)d_in[5];
    const float* qkv_w  = (const float*)d_in[6];
    const float* qkv_b  = (const float*)d_in[7];
    const float* proj_w = (const float*)d_in[8];
    const float* proj_b = (const float*)d_in[9];
    const float* ln2_w  = (const float*)d_in[10];
    const float* ln2_b  = (const float*)d_in[11];
    const float* fc1_w  = (const float*)d_in[12];
    const float* fc1_b  = (const float*)d_in[13];
    const float* fc2_w  = (const float*)d_in[14];
    const float* fc2_b  = (const float*)d_in[15];
    const float* lnf_w  = (const float*)d_in[16];
    const float* lnf_b  = (const float*)d_in[17];

    char* ws = (char*)d_ws;
    size_t off = 0;
    auto take = [&](size_t bytes) -> char* {
        char* p = ws + off;
        off += (bytes + 255) & ~(size_t)255;
        return p;
    };
    bf16_t* xp    = (bf16_t*)take((size_t)ROWS * D_MODEL * 2);
    float*  hbuf  = (float*) take((size_t)ROWS * D_MODEL * 4);
    bf16_t* ybuf  = (bf16_t*)take((size_t)ROWS * D_MODEL * 2);
    bf16_t* qkvb  = (bf16_t*)take((size_t)ROWS * QKV3 * 2);
    bf16_t* obuf  = (bf16_t*)take((size_t)ROWS * D_MODEL * 2);
    bf16_t* hid   = (bf16_t*)take((size_t)ROWS * HID_DIM * 2);
    bf16_t* convT = (bf16_t*)take((size_t)D_MODEL * D_MODEL * 2);
    bf16_t* qkvT  = (bf16_t*)take((size_t)LAYERS * D_MODEL * QKV3 * 2);
    bf16_t* projT = (bf16_t*)take((size_t)LAYERS * D_MODEL * D_MODEL * 2);
    bf16_t* fc1T  = (bf16_t*)take((size_t)LAYERS * D_MODEL * HID_DIM * 2);
    bf16_t* fc2T  = (bf16_t*)take((size_t)LAYERS * HID_DIM * D_MODEL * 2);

    const int ew = 256;
    // one-time weight conversion/transposition to bf16 [N][K]
    cvt_bf16_k<<<(D_MODEL * D_MODEL + ew - 1) / ew, ew, 0, stream>>>(
        conv_w, convT, D_MODEL * D_MODEL);
    for (int l = 0; l < LAYERS; ++l) {
        wtrans_cvt_k<<<(D_MODEL * QKV3 + ew - 1) / ew, ew, 0, stream>>>(
            qkv_w + (size_t)l * D_MODEL * QKV3, qkvT + (size_t)l * QKV3 * D_MODEL,
            D_MODEL, QKV3);
        wtrans_cvt_k<<<(D_MODEL * D_MODEL + ew - 1) / ew, ew, 0, stream>>>(
            proj_w + (size_t)l * D_MODEL * D_MODEL, projT + (size_t)l * D_MODEL * D_MODEL,
            D_MODEL, D_MODEL);
        wtrans_cvt_k<<<(D_MODEL * HID_DIM + ew - 1) / ew, ew, 0, stream>>>(
            fc1_w + (size_t)l * D_MODEL * HID_DIM, fc1T + (size_t)l * HID_DIM * D_MODEL,
            D_MODEL, HID_DIM);
        wtrans_cvt_k<<<(HID_DIM * D_MODEL + ew - 1) / ew, ew, 0, stream>>>(
            fc2_w + (size_t)l * HID_DIM * D_MODEL, fc2T + (size_t)l * D_MODEL * HID_DIM,
            HID_DIM, D_MODEL);
    }

    patch_gather_k<<<(ROWS * D_MODEL + ew - 1) / ew, ew, 0, stream>>>(x, xp);
    gemm_wmma<0><<<dim3(D_MODEL / 128, ROWS / 128), 256, 0, stream>>>(
        xp, convT, conv_b, hbuf, nullptr, ROWS, D_MODEL, D_MODEL);
    add_pos_k<<<(ROWS * D_MODEL + ew - 1) / ew, ew, 0, stream>>>(hbuf, pos);

    for (int l = 0; l < LAYERS; ++l) {
        layernorm_k<true><<<ROWS, 256, 0, stream>>>(
            hbuf, ln1_w + l * D_MODEL, ln1_b + l * D_MODEL, ybuf, nullptr);
        gemm_wmma<2><<<dim3(QKV3 / 128, ROWS / 128), 256, 0, stream>>>(
            ybuf, qkvT + (size_t)l * QKV3 * D_MODEL, qkv_b + (size_t)l * QKV3,
            nullptr, qkvb, ROWS, QKV3, D_MODEL);
        attn_k<<<BATCH * NHEAD, 128, 0, stream>>>(qkvb, obuf);
        gemm_wmma<1><<<dim3(D_MODEL / 128, ROWS / 128), 256, 0, stream>>>(
            obuf, projT + (size_t)l * D_MODEL * D_MODEL, proj_b + (size_t)l * D_MODEL,
            hbuf, nullptr, ROWS, D_MODEL, D_MODEL);
        layernorm_k<true><<<ROWS, 256, 0, stream>>>(
            hbuf, ln2_w + l * D_MODEL, ln2_b + l * D_MODEL, ybuf, nullptr);
        gemm_wmma<3><<<dim3(HID_DIM / 128, ROWS / 128), 256, 0, stream>>>(
            ybuf, fc1T + (size_t)l * HID_DIM * D_MODEL, fc1_b + (size_t)l * HID_DIM,
            nullptr, hid, ROWS, HID_DIM, D_MODEL);
        gemm_wmma<1><<<dim3(D_MODEL / 128, ROWS / 128), 256, 0, stream>>>(
            hid, fc2T + (size_t)l * D_MODEL * HID_DIM, fc2_b + (size_t)l * D_MODEL,
            hbuf, nullptr, ROWS, D_MODEL, HID_DIM);
    }
    layernorm_k<false><<<ROWS, 256, 0, stream>>>(hbuf, lnf_w, lnf_b, nullptr, (float*)d_out);
}
